// DynamicTokenRouter_66305705116355
// MI455X (gfx1250) — matrix operand order repaired and checked by reading, not compiled, not run on hardware
//
#include <hip/hip_runtime.h>
#include <hip/hip_bf16.h>
#include <math.h>

// ---------------------------------------------------------------------------
// DynamicTokenRouter for MI455X (gfx1250, wave32, WMMA + TDM)
//   tokens [4,8192,1024] f32 -> scores via bf16 WMMA MLP -> radix top-k ->
//   gather rows + mask.
// ---------------------------------------------------------------------------

#define DIM     1024
#define HDIM    512
#define NTOK    8192
#define NKEEP   4096
#define BATCH   4

typedef __attribute__((ext_vector_type(16))) __bf16 v16bf;
typedef __attribute__((ext_vector_type(8)))  __bf16 v8bf;
typedef __attribute__((ext_vector_type(4)))  __bf16 v4bf;
typedef __attribute__((ext_vector_type(8)))  float  v8f;
typedef __attribute__((ext_vector_type(4)))  float  v4f;

typedef __attribute__((ext_vector_type(4))) unsigned int u32x4;
typedef __attribute__((ext_vector_type(8))) int          i32x8;
typedef __attribute__((ext_vector_type(4))) int          i32x4;

#if __has_builtin(__builtin_amdgcn_tensor_load_to_lds) && \
    __has_builtin(__builtin_amdgcn_s_wait_tensorcnt)
#define USE_TDM 1
#else
#define USE_TDM 0
#endif

// ---------------------------------------------------------------------------
// Kernel 1: repack W1 [1024,512] f32 (row-major d,h) into bf16 tiles laid out
// in the k order this file's WMMA operands use (k contiguous per lane half):
//   tile (ntile in 0..31 over H, ktile in 0..31 over D):
//     lane l holds N = ntile*16 + (l&15), K = ktile*32 + 16*(l>=16) + e, e=0..15
//   stored contiguously: W1p[((ntile*32 + ktile)*32 + lane)*16 + e]
// so the score kernel loads one v16bf (32B contiguous) per lane per tile.
// ---------------------------------------------------------------------------
__global__ __launch_bounds__(32)
void pack_w1_kernel(const float* __restrict__ W1, __bf16* __restrict__ W1p) {
  const int ntile = blockIdx.x;      // 0..31
  const int ktile = blockIdx.y;      // 0..31
  const int lane  = threadIdx.x;     // 0..31
  const int n     = ntile * 16 + (lane & 15);
  const int kbase = ktile * 32 + ((lane >> 4) << 4);
  __bf16* dst = W1p + (((size_t)(ntile * 32 + ktile) * 32 + lane) * 16);
#pragma unroll
  for (int e = 0; e < 16; ++e)
    dst[e] = (__bf16)W1[(size_t)(kbase + e) * HDIM + n];
}

// ---------------------------------------------------------------------------
// Kernel 2: scores. Block = 16 tokens, 128 threads (4 waves).
// Token tile (16 rows x 1024 f32 = 64KB) is DMA'd into LDS by the Tensor Data
// Mover (tensor_load_to_lds, TENSORcnt), then converted to the bf16 WMMA-A
// layout in LDS. Wave w owns H-columns [w*128, w*128+128) as 8 N-tiles of 16.
// A fragment (16x32 bf16) layout loaded here (same k order as W1p):
//   lane l: row M = l&15; elems 0..7 = K k0 + 16*(l>=16) + {0..7};
//                        elems 8..15 = K k0 + 8 + 16*(l>=16) + {0..7}
// ---------------------------------------------------------------------------
__global__ __launch_bounds__(128)
void score_kernel(const float* __restrict__ tokens,
                  const __bf16* __restrict__ W1p,
                  const float* __restrict__ b1,
                  const float* __restrict__ W2,
                  const float* __restrict__ b2,
                  float* __restrict__ scores) {
  constexpr int LDA = DIM + 8;                 // padded bf16 row stride (16B-aligned)
  __shared__ float  ldsF[16 * DIM];            // raw fp32 rows (TDM destination)
  __shared__ __bf16 ldsA[16 * LDA];            // bf16 WMMA-A staging
  __shared__ float  ldsS[16];

  const int b    = blockIdx.y;
  const int m0   = blockIdx.x * 16;
  const int tid  = threadIdx.x;
  const int wave = tid >> 5;
  const int lane = tid & 31;

#if USE_TDM
  // ---- TDM: DMA 16 x 1024 f32 tile (contiguous 64KB) global -> LDS ----
  if (wave == 0) {
    const float* gsrc = tokens + ((size_t)b * NTOK + m0) * DIM;
    unsigned long long gaddr = (unsigned long long)gsrc;
    // ISA 10.2: low 32 bits of a flat LDS address are the LDS byte offset.
    unsigned lds_off = (unsigned)(unsigned long long)(&ldsF[0]);

    // D# group 0: count=1 | lds_addr | global_addr[31:0] | global_addr[56:32],type=2
    u32x4 g0 = { 1u,
                 lds_off,
                 (unsigned)(gaddr & 0xffffffffu),
                 (unsigned)((gaddr >> 32) & 0x01ffffffu) | (2u << 30) };
    // D# group 1 (bit offsets within 256b):
    //  [17:16] data_size=2 (4B)  [79:48] tensor_dim0=1024
    //  [111:80] tensor_dim1=8192 [127:112] tile_dim0=1024
    //  [143:128] tile_dim1=16    [207:160] tensor_dim0_stride=1024
    //  workgroup_mask=0 (not in a cluster), no pad/iterate/barrier
    i32x8 g1 = { (int)(2u << 16),
                 (int)(1024u << 16),
                 (int)(8192u << 16),
                 (int)(1024u << 16),
                 16,
                 1024,
                 0, 0 };
    i32x4 gz = { 0, 0, 0, 0 };
#if defined(__clang_major__) && (__clang_major__ >= 23)
    i32x8 gz8 = { 0, 0, 0, 0, 0, 0, 0, 0 };
    __builtin_amdgcn_tensor_load_to_lds(g0, g1, gz, gz, gz8, 0);
#else
    __builtin_amdgcn_tensor_load_to_lds(g0, g1, gz, gz, 0);
#endif
    __builtin_amdgcn_s_wait_tensorcnt(0);      // s_wait_tensorcnt 0
  }
  if (tid < 16) ldsS[tid] = 0.0f;
  __syncthreads();                             // publish TDM tile to all waves

  // ---- convert fp32 tile -> bf16 WMMA-A layout (LDS -> LDS) ----
  for (int i = tid; i < 16 * (DIM / 4); i += 128) {
    const int row = i / (DIM / 4);
    const int c4  = i % (DIM / 4);
    v4f f = *(const v4f*)(&ldsF[row * DIM + c4 * 4]);
    v4bf h;
    h[0] = (__bf16)f[0]; h[1] = (__bf16)f[1];
    h[2] = (__bf16)f[2]; h[3] = (__bf16)f[3];
    *(v4bf*)(&ldsA[row * LDA + c4 * 4]) = h;
  }
  __syncthreads();
#else
  // ---- fallback staging: global float4 loads -> bf16 LDS ----
  {
    const float* src = tokens + ((size_t)b * NTOK + m0) * DIM;
    for (int i = tid; i < 16 * (DIM / 4); i += 128) {
      const int row = i / (DIM / 4);
      const int c4  = i % (DIM / 4);
      v4f f = *(const v4f*)(src + (size_t)row * DIM + c4 * 4);
      v4bf h;
      h[0] = (__bf16)f[0]; h[1] = (__bf16)f[1];
      h[2] = (__bf16)f[2]; h[3] = (__bf16)f[3];
      *(v4bf*)(&ldsA[row * LDA + c4 * 4]) = h;
    }
    if (tid < 16) ldsS[tid] = 0.0f;
  }
  __syncthreads();
#endif

  // ---- WMMA main loop: 8 accumulators (one per N-tile), K in steps of 32 ----
  v8f acc[8];
#pragma unroll
  for (int t = 0; t < 8; ++t) { v8f z = {}; acc[t] = z; }

  const int row  = lane & 15;
  const int aoff = (lane >> 4) * 16;

  for (int kt = 0; kt < DIM / 32; ++kt) {
    const int k0 = kt * 32;
    v16bf aval;
    v8bf a_lo = *(const v8bf*)(&ldsA[row * LDA + k0 + aoff]);        // ds_load_b128
    v8bf a_hi = *(const v8bf*)(&ldsA[row * LDA + k0 + 8 + aoff]);    // ds_load_b128
#pragma unroll
    for (int e = 0; e < 8; ++e) { aval[e] = a_lo[e]; aval[e + 8] = a_hi[e]; }

    // prefetch next K-tile of B for this wave's first N-tile
    if (kt + 1 < DIM / 32) {
      const __bf16* pf = W1p +
          ((((size_t)(wave * 8) * 32 + (kt + 1)) * 32 + lane) * 16);
      __builtin_prefetch((const void*)pf, 0, 1);                     // global_prefetch_b8
    }

#pragma unroll
    for (int t = 0; t < 8; ++t) {
      const __bf16* bp = W1p +
          ((((size_t)(wave * 8 + t) * 32 + kt) * 32 + lane) * 16);
      v16bf bval = *(const v16bf*)bp;                                // 2x global_load_b128
      acc[t] = __builtin_amdgcn_wmma_f32_16x16x32_bf16(
          /*neg_a=*/false, aval, /*neg_b=*/false, bval,
          /*c_mod=*/(short)0, acc[t], /*reuse_a=*/false, /*reuse_b=*/false);
    }
  }

  // ---- epilogue: +b1, exact GELU, *W2, reduce over N ----
  // C layout: VGPR r, lanes 0-15 -> (M=r,   N=lane);
  //                  lanes 16-31 -> (M=r+8, N=lane-16)
  float tokAcc[8];
#pragma unroll
  for (int r = 0; r < 8; ++r) tokAcc[r] = 0.0f;

#pragma unroll
  for (int t = 0; t < 8; ++t) {
    const int n   = wave * 128 + t * 16 + (lane & 15);
    const float bn = b1[n];
    const float wv = W2[n];
#pragma unroll
    for (int r = 0; r < 8; ++r) {
      float x = acc[t][r] + bn;
      float g = 0.5f * x * (1.0f + erff(x * 0.70710678118654752f)); // exact GELU
      tokAcc[r] += g * wv;
    }
  }

  // reduce across the 16-lane N-groups (halves of the wave stay disjoint)
#pragma unroll
  for (int r = 0; r < 8; ++r) {
    float v = tokAcc[r];
    v += __shfl_xor(v, 1, 32);
    v += __shfl_xor(v, 2, 32);
    v += __shfl_xor(v, 4, 32);
    v += __shfl_xor(v, 8, 32);
    if ((lane & 15) == 0)
      atomicAdd(&ldsS[r + ((lane >> 4) << 3)], v);   // token r (lo) / r+8 (hi)
  }
  __syncthreads();

  if (tid < 16)
    scores[(size_t)b * NTOK + m0 + tid] = ldsS[tid] + b2[0];
}

// ---------------------------------------------------------------------------
// Kernel 3: per-batch top-NKEEP via 4-pass MSB radix select + ordered compact.
// One 1024-thread block per batch; thread t owns tokens [t*8, t*8+8).
// Tie-break = lowest token index first (matches jax.lax.top_k).
// ---------------------------------------------------------------------------
__global__ __launch_bounds__(1024)
void select_kernel(const float* __restrict__ scores,
                   float* __restrict__ mask_out,
                   int* __restrict__ idx_out) {
  __shared__ unsigned int keys[NTOK];
  __shared__ unsigned int hist[256];
  __shared__ unsigned int scn[1024];
  __shared__ unsigned int bc_prefix, bc_k;

  const int b   = blockIdx.x;
  const int tid = threadIdx.x;

  // monotone float -> uint key (larger float => larger key)
  for (int i = tid; i < NTOK; i += 1024) {
    unsigned int u = __float_as_uint(scores[(size_t)b * NTOK + i]);
    u = (u & 0x80000000u) ? ~u : (u | 0x80000000u);
    keys[i] = u;
  }
  if (tid == 0) { bc_prefix = 0u; bc_k = NKEEP; }
  __syncthreads();

  // 4 radix passes, MSB first, to find the NKEEP-th largest key
  for (int pass = 0; pass < 4; ++pass) {
    const int shift = 24 - pass * 8;
    if (tid < 256) hist[tid] = 0u;
    __syncthreads();
    const unsigned int pfx = bc_prefix;
    const unsigned int msk = (pass == 0) ? 0u : (0xFFFFFFFFu << (32 - pass * 8));
    for (int i = tid; i < NTOK; i += 1024) {
      unsigned int u = keys[i];
      if ((u & msk) == pfx)
        atomicAdd(&hist[(u >> shift) & 255u], 1u);
    }
    __syncthreads();
    if (tid == 0) {
      unsigned int rem = bc_k;
      int d = 255;
      while (d > 0) {
        unsigned int c = hist[d];
        if (rem <= c) break;
        rem -= c;
        --d;
      }
      bc_prefix |= ((unsigned int)d << shift);
      bc_k = rem;
    }
    __syncthreads();
  }
  const unsigned int T = bc_prefix;   // key of the NKEEP-th largest element

  // per-thread counts over its 8 contiguous tokens, packed (gt<<16 | eq)
  unsigned int gt = 0, eq = 0;
  const int base = tid * 8;
#pragma unroll
  for (int i = 0; i < 8; ++i) {
    unsigned int u = keys[base + i];
    gt += (u > T);
    eq += (u == T);
  }
  scn[tid] = (gt << 16) | eq;
  __syncthreads();

  // Hillis-Steele inclusive scan over 1024 packed counts
  for (int off = 1; off < 1024; off <<= 1) {
    unsigned int v = (tid >= off) ? scn[tid - off] : 0u;
    __syncthreads();
    scn[tid] += v;
    __syncthreads();
  }
  const unsigned int total    = scn[1023];
  const unsigned int incl     = scn[tid];
  unsigned int g = (incl >> 16) - gt;        // gt tokens before my chunk
  unsigned int e = (incl & 0xFFFFu) - eq;    // eq tokens before my chunk
  const unsigned int total_gt = total >> 16;
  const unsigned int need_eq  = NKEEP - total_gt;

  // ordered walk: emit mask + compacted (index-sorted) keep list
#pragma unroll
  for (int i = 0; i < 8; ++i) {
    const int token = base + i;
    const unsigned int u = keys[token];
    bool keep = false;
    unsigned int pos = 0;
    if (u > T) {
      keep = true;
      pos = g + ((e < need_eq) ? e : need_eq);
      ++g;
    } else if (u == T) {
      if (e < need_eq) { keep = true; pos = g + e; }
      ++e;
    }
    mask_out[(size_t)b * NTOK + token] = keep ? 1.0f : 0.0f;
    if (keep) idx_out[(size_t)b * NKEEP + pos] = token;
  }
}

// ---------------------------------------------------------------------------
// Kernel 4: gather kept rows. One block per output row; 256 threads x float4
// = one contiguous 4KB row (b128 in / b128 out, fully coalesced).
// ---------------------------------------------------------------------------
__global__ __launch_bounds__(256)
void gather_kernel(const float* __restrict__ tokens,
                   const int* __restrict__ idx,
                   float* __restrict__ selected) {
  const int b = blockIdx.y;
  const int k = blockIdx.x;
  const int row = idx[(size_t)b * NKEEP + k];
  const v4f* src = (const v4f*)(tokens + ((size_t)b * NTOK + row) * DIM);
  v4f* dst = (v4f*)(selected + ((size_t)b * NKEEP + (size_t)k) * DIM);
  dst[threadIdx.x] = src[threadIdx.x];
}

// ---------------------------------------------------------------------------
extern "C" void kernel_launch(void* const* d_in, const int* in_sizes, int n_in,
                              void* d_out, int out_size, void* d_ws, size_t ws_size,
                              hipStream_t stream) {
  const float* tokens = (const float*)d_in[0];
  const float* W1     = (const float*)d_in[1];
  const float* b1     = (const float*)d_in[2];
  const float* W2     = (const float*)d_in[3];
  const float* b2     = (const float*)d_in[4];

  float* selected = (float*)d_out;                                   // [4,4096,1024]
  float* mask     = (float*)d_out + (size_t)BATCH * NKEEP * DIM;     // [4,8192]

  // workspace layout
  __bf16* W1p    = (__bf16*)d_ws;                                    // 1 MiB
  float*  scores = (float*)((char*)d_ws + (size_t)DIM * HDIM * 2);   // 128 KiB
  int*    idxbuf = (int*)((char*)d_ws + (size_t)DIM * HDIM * 2
                                      + (size_t)BATCH * NTOK * 4);   // 64 KiB

  pack_w1_kernel<<<dim3(HDIM / 16, DIM / 32), 32, 0, stream>>>(W1, W1p);
  score_kernel<<<dim3(NTOK / 16, BATCH), 128, 0, stream>>>(tokens, W1p, b1, W2, b2, scores);
  select_kernel<<<BATCH, 1024, 0, stream>>>(scores, mask, idxbuf);
  gather_kernel<<<dim3(NKEEP, BATCH), 256, 0, stream>>>(tokens, idxbuf, selected);
}
